// Net_47442208751911
// MI455X (gfx1250) — matrix operand order repaired and checked by reading, not compiled, not run on hardware
//
#include <hip/hip_runtime.h>
#include <hip/hip_bf16.h>

// ---------------------------------------------------------------------------
// Fused 2-layer LSTM for MI455X (gfx1250, wave32, WMMA).
//
// ~210 GFLOP of [*,256]x[256,1024] GEMMs chained over T=256 steps; weights
// (~1.6MB f16) are L2-resident, so the kernel is compute/latency bound.
//  - batch rows are recurrence-independent -> 32 persistent WGs x 16 rows,
//    each runs all 256 timesteps of BOTH layers fused (h1 never touches HBM).
//  - h0/h1 state in LDS (f16, WMMA-A fragment order); c0/c1 in VGPRs.
//  - wave w owns gate columns [16w,16w+16) of ALL FOUR gates, so the LSTM
//    cell update is done fully in registers from the WMMA accumulators.
//  - a laundered zero OFFSET (not pointer!) defeats LICM of the ~100 weight
//    B-fragments (no scratch spill) while keeping global (AS1) addressing:
//    global_load_b128 saddr+imm instead of flat_load + 64-bit VALU adds.
// ---------------------------------------------------------------------------

typedef __attribute__((ext_vector_type(16))) _Float16 v16h;
typedef __attribute__((ext_vector_type(8)))  float    v8f;

#define Bsz 512
#define Tsz 256
#define Isz 10
#define Hsz 256
#define Gsz 1024          // 4*H
#define BM  16            // batch rows per workgroup
#define THREADS 512       // 16 waves
#define HS  272           // padded h row stride (halves) -> rotates LDS banks

// A-fragment order within each 32-wide K chunk (identity, contiguous):
// lanes 0-15 supply K {0..15}, lanes 16-31 supply K {16..31}.
__device__ __host__ inline int slot32(int k) {
    return k;
}
__device__ inline int unslot32(int s) {
    return s;
}

__device__ inline float fast_sigmoid(float x) {
    return 1.0f / (1.0f + __expf(-x));
}
__device__ inline float fast_tanh(float x) {
    float e = __expf(2.0f * x);
    return (e - 1.0f) / (e + 1.0f);
}

// -------------------------- prep: weights to f16 ---------------------------
__global__ void prep_weights(const float* __restrict__ Wih0, const float* __restrict__ Whh0,
                             const float* __restrict__ bih0, const float* __restrict__ bhh0,
                             const float* __restrict__ Wih1, const float* __restrict__ Whh1,
                             const float* __restrict__ bih1, const float* __restrict__ bhh1,
                             _Float16* __restrict__ whh0f, _Float16* __restrict__ wih1f,
                             _Float16* __restrict__ whh1f, _Float16* __restrict__ wih0p,
                             float* __restrict__ b0, float* __restrict__ b1) {
    int i = blockIdx.x * 256 + threadIdx.x;
    const int NW = Gsz * Hsz;                       // 262144
    if (i < NW) { whh0f[i] = (_Float16)Whh0[i]; return; }
    i -= NW;
    if (i < NW) { wih1f[i] = (_Float16)Wih1[i]; return; }
    i -= NW;
    if (i < NW) { whh1f[i] = (_Float16)Whh1[i]; return; }
    i -= NW;
    if (i < Gsz * 32) {                             // zero-padded K=10 -> 32
        int n = i >> 5, s = i & 31;
        wih0p[i] = (s < Isz) ? (_Float16)Wih0[n * Isz + s] : (_Float16)0.0f;
        return;
    }
    i -= Gsz * 32;
    if (i < Gsz) { b0[i] = bih0[i] + bhh0[i]; return; }
    i -= Gsz;
    if (i < Gsz) { b1[i] = bih1[i] + bhh1[i]; return; }
}

// ------------------- prep: x -> f16, K padded + A-permuted -----------------
__global__ void pack_x(const float* __restrict__ x, _Float16* __restrict__ xp) {
    long i = (long)blockIdx.x * 256 + threadIdx.x;   // B*T*32 = 4194304 exactly
    int s = (int)(i & 31);
    long bt = i >> 5;                                // b*T + t
    int k = unslot32(s);
    float v = (k < Isz) ? x[bt * Isz + k] : 0.0f;
    xp[i] = (_Float16)v;
}

// ----------------------------- fused LSTM ----------------------------------
__launch_bounds__(THREADS, 1)
__global__ void lstm_fused(const _Float16* __restrict__ whh0f_in,
                           const _Float16* __restrict__ wih1f_in,
                           const _Float16* __restrict__ whh1f_in,
                           const _Float16* __restrict__ wih0p,
                           const float* __restrict__ b0g,
                           const float* __restrict__ b1g,
                           const _Float16* __restrict__ xp,
                           const float* __restrict__ Wlin,
                           const float* __restrict__ blin,
                           float* __restrict__ out) {
    extern __shared__ char smem[];
    _Float16* h0p = (_Float16*)smem;                       // [BM][HS] permuted f16
    _Float16* h1p = (_Float16*)(smem + BM * HS * 2);

    const int tid  = threadIdx.x;
    const int lane = tid & 31;
    const int w    = tid >> 5;       // wave 0..15
    const int m    = lane & 15;      // M-row (A) / N-col (B,C) within tile
    const int half = lane >> 4;      // K-half select
    const int b0r  = blockIdx.x * BM;

    // zero h state
    for (int i = tid; i < BM * HS; i += THREADS) {
        h0p[i] = (_Float16)0.0f; h1p[i] = (_Float16)0.0f;
    }

    // Wave w owns gate columns [16w, 16w+16) of each of the 4 gates.
    // Per-lane owned column within gate g: ncol(g) = g*256 + w*16 + m.
    const int jcol = w * 16 + m;                     // hidden index this lane updates
    const int hslot = (jcol >> 5) * 32 + slot32(jcol & 31);  // permuted in-row offset

    // loop-invariant per-lane biases (4 per layer)
    const float bi0 = b0g[jcol], bf0 = b0g[jcol + 256], bg0 = b0g[jcol + 512], bo0 = b0g[jcol + 768];
    const float bi1 = b1g[jcol], bf1 = b1g[jcol + 256], bg1 = b1g[jcol + 512], bo1 = b1g[jcol + 768];

    // register-cached input-weight fragments (small: 4 x v16h = 32 VGPRs)
    v16h bx[4];
#pragma unroll
    for (int g = 0; g < 4; ++g)
        bx[g] = *(const v16h*)(wih0p + (size_t)(g * 256 + jcol) * 32 + half * 16);

    // cell state in registers: rows (half*8 + r), col jcol
    float c0r[8], c1r[8];
#pragma unroll
    for (int r = 0; r < 8; ++r) { c0r[r] = 0.0f; c1r[r] = 0.0f; }

    const _Float16* xrow = xp + ((size_t)(b0r + m)) * Tsz * 32 + half * 16;

    // per-lane base element offset into the weight matrices (column jcol, K-half)
    const size_t wbase = (size_t)jcol * Hsz + half * 16;

    __syncthreads();

    for (int t = 0; t < Tsz; ++t) {
        // Launder a ZERO OFFSET each iteration: addresses become loop-variant
        // (defeats LICM -> no 800-VGPR hoist/spill) while the base pointers
        // keep kernarg provenance -> global_load_b128 saddr addressing.
        size_t z0 = 0, z1 = 0, z2 = 0;
        asm volatile("" : "+s"(z0), "+s"(z1), "+s"(z2));
        const _Float16* const whh0f = whh0f_in + z0;
        const _Float16* const wih1f = wih1f_in + z1;
        const _Float16* const whh1f = whh1f_in + z2;

        if (t + 1 < Tsz) __builtin_prefetch(xrow + (size_t)(t + 1) * 32, 0, 3);

        // ================= layer 0 gates: x_t*Wih0' + h0*Whh0' =============
        v8f acc[4] = {};
        {
            const v16h ax = *(const v16h*)(xrow + (size_t)t * 32);
#pragma unroll
            for (int g = 0; g < 4; ++g)
                acc[g] = __builtin_amdgcn_wmma_f32_16x16x32_f16(
                    false, ax, false, bx[g], (short)0, acc[g], false, false);
#pragma unroll
            for (int kc = 0; kc < 8; ++kc) {
                const v16h ah = *(const v16h*)(h0p + m * HS + kc * 32 + half * 16);
#pragma unroll
                for (int g = 0; g < 4; ++g) {
                    const v16h bw = *(const v16h*)(whh0f + wbase + (size_t)g * 256 * Hsz + kc * 32);
                    acc[g] = __builtin_amdgcn_wmma_f32_16x16x32_f16(
                        false, ah, false, bw, (short)0, acc[g], false, false);
                }
            }
        }
        __syncthreads();   // all waves done reading old h0p

        // ------- layer 0 cell update, fully in registers -> h0p, c0 --------
#pragma unroll
        for (int r = 0; r < 8; ++r) {
            float gi = fast_sigmoid(acc[0][r] + bi0);
            float gf = fast_sigmoid(acc[1][r] + bf0);
            float gg = fast_tanh   (acc[2][r] + bg0);
            float go = fast_sigmoid(acc[3][r] + bo0);
            float c  = gf * c0r[r] + gi * gg;
            c0r[r] = c;
            h0p[(r + 8 * half) * HS + hslot] = (_Float16)(go * fast_tanh(c));
        }
        __syncthreads();   // new h0p visible to all waves

        // ============ layer 1 gates: h0_new*Wih1' + h1*Whh1' ===============
        v8f acc1[4] = {};
        {
#pragma unroll
            for (int kc = 0; kc < 8; ++kc) {
                const v16h ah = *(const v16h*)(h0p + m * HS + kc * 32 + half * 16);
#pragma unroll
                for (int g = 0; g < 4; ++g) {
                    const v16h bw = *(const v16h*)(wih1f + wbase + (size_t)g * 256 * Hsz + kc * 32);
                    acc1[g] = __builtin_amdgcn_wmma_f32_16x16x32_f16(
                        false, ah, false, bw, (short)0, acc1[g], false, false);
                }
            }
#pragma unroll
            for (int kc = 0; kc < 8; ++kc) {
                const v16h ah = *(const v16h*)(h1p + m * HS + kc * 32 + half * 16);
#pragma unroll
                for (int g = 0; g < 4; ++g) {
                    const v16h bw = *(const v16h*)(whh1f + wbase + (size_t)g * 256 * Hsz + kc * 32);
                    acc1[g] = __builtin_amdgcn_wmma_f32_16x16x32_f16(
                        false, ah, false, bw, (short)0, acc1[g], false, false);
                }
            }
        }
        __syncthreads();   // all waves done reading old h1p

        // ------- layer 1 cell update, fully in registers -> h1p, c1 --------
#pragma unroll
        for (int r = 0; r < 8; ++r) {
            float gi = fast_sigmoid(acc1[0][r] + bi1);
            float gf = fast_sigmoid(acc1[1][r] + bf1);
            float gg = fast_tanh   (acc1[2][r] + bg1);
            float go = fast_sigmoid(acc1[3][r] + bo1);
            float c  = gf * c1r[r] + gi * gg;
            c1r[r] = c;
            h1p[(r + 8 * half) * HS + hslot] = (_Float16)(go * fast_tanh(c));
        }
        __syncthreads();   // new h1p visible for next step
    }

    // ----------------- final linear head: y = h1[:,T-1] @ Wlin' + b --------
    if (tid < BM) {
        float acc = blin[0];
        for (int s = 0; s < Hsz; ++s) {
            int j = (s >> 5) * 32 + unslot32(s & 31);   // undo A-permutation
            acc += (float)h1p[tid * HS + s] * Wlin[j];
        }
        out[b0r + tid] = acc;
    }
}

// ---------------------------------------------------------------------------
extern "C" void kernel_launch(void* const* d_in, const int* in_sizes, int n_in,
                              void* d_out, int out_size, void* d_ws, size_t ws_size,
                              hipStream_t stream) {
    const float* x    = (const float*)d_in[0];
    const float* Wih0 = (const float*)d_in[1];
    const float* Whh0 = (const float*)d_in[2];
    const float* bih0 = (const float*)d_in[3];
    const float* bhh0 = (const float*)d_in[4];
    const float* Wih1 = (const float*)d_in[5];
    const float* Whh1 = (const float*)d_in[6];
    const float* bih1 = (const float*)d_in[7];
    const float* bhh1 = (const float*)d_in[8];
    const float* Wlin = (const float*)d_in[9];
    const float* blin = (const float*)d_in[10];

    char* ws = (char*)d_ws;
    _Float16* whh0f = (_Float16*)(ws + 0);          // 512 KB
    _Float16* wih1f = (_Float16*)(ws + 524288);     // 512 KB
    _Float16* whh1f = (_Float16*)(ws + 1048576);    // 512 KB
    _Float16* wih0p = (_Float16*)(ws + 1572864);    // 64 KB
    float*    b0    = (float*)   (ws + 1638400);    // 4 KB
    float*    b1    = (float*)   (ws + 1642496);    // 4 KB
    _Float16* xp    = (_Float16*)(ws + 1646592);    // 8 MB

    // 3*262144 + 32768 + 2048 = 821248 = 3208 * 256
    prep_weights<<<3208, 256, 0, stream>>>(Wih0, Whh0, bih0, bhh0,
                                           Wih1, Whh1, bih1, bhh1,
                                           whh0f, wih1f, whh1f, wih0p, b0, b1);
    // 512*256*32 = 4194304 = 16384 * 256
    pack_x<<<16384, 256, 0, stream>>>(x, xp);

    const size_t lds = 2 * (size_t)BM * HS * 2;     // 17408 B
    lstm_fused<<<Bsz / BM, THREADS, lds, stream>>>(whh0f, wih1f, whh1f, wih0p,
                                                   b0, b1, xp, Wlin, blin,
                                                   (float*)d_out);
}
